// FusedExpertsNetwork_30545807409459
// MI455X (gfx1250) — matrix operand order repaired and checked by reading, not compiled, not run on hardware
//
#include <hip/hip_runtime.h>
#include <hip/hip_bf16.h>

// ---------------------------------------------------------------------------
// Fused expert MLP: y = relu(x @ W1^T + b1) @ W2 + b2, batched over E experts.
// E=8, C=1024, M=2048, H=8192, O=2048.  fp32 in/out, bf16 WMMA compute,
// bf16 intermediate activations in d_ws (E*C*H*2 = 128 MB).
//
// Software-pipelined: global loads for tile k+1 are issued into registers
// right after the fill barrier, overlapping L2/HBM latency with the 16 WMMAs
// of tile k.  Per-expert weight working set (~72 MB) fits in the 192 MB L2,
// so HBM traffic stays compulsory (~1.15 GB) and the kernel is WMMA-bound.
// ---------------------------------------------------------------------------

#define NE 8
#define NC 1024
#define NM 2048
#define NH 8192
#define NO 2048

#define TM 128          // tile rows (tokens c)
#define TN 128          // tile cols (h or o)
#define TK 64           // k-chunk staged in LDS per iteration
#define KS 32           // k per WMMA instruction (bf16)
#define LDT 72          // LDS row stride in bf16 elements (TK + 8 pad -> 144 B, 16B multiple)

typedef __attribute__((ext_vector_type(16))) __bf16   v16bf;
typedef __attribute__((ext_vector_type(8)))  float    v8f;
typedef __attribute__((ext_vector_type(2)))  unsigned uint2v;
typedef __attribute__((ext_vector_type(4)))  unsigned uint4v;
typedef __attribute__((ext_vector_type(8)))  unsigned uint8v;

// fp32 -> bf16, round to nearest even
static __device__ __forceinline__ unsigned f2bf1(float f) {
    unsigned u = __builtin_bit_cast(unsigned, f);
    return (u + 0x7fffu + ((u >> 16) & 1u)) >> 16;
}
static __device__ __forceinline__ unsigned packbf(float a, float b) {
    return (f2bf1(b) << 16) | (f2bf1(a) & 0xffffu);
}

// A-fragment (16x32 bf16, M x K): lane m = lane&15; lane half selects K group
// {0-7,16-23} (lanes 0-15) vs {8-15,24-31} (lanes 16-31).  Row-major [row][k]
// LDS tile -> two 16-byte chunks at +0 / +32 (+16 for upper lane half).
static __device__ __forceinline__ v16bf ldfragA(const unsigned short* rowbase, int lane) {
    const char* p = (const char*)rowbase + ((lane >> 4) << 4);
    uint8v t;
    t.lo = *(const uint4v*)(p);
    t.hi = *(const uint4v*)(p + 32);
    return __builtin_bit_cast(v16bf, t);
}
// B-fragment (32x16 bf16, K x N): lane n = lane&15; lanes 0-15 hold K={0-7,16-23},
// lanes 16-31 hold K={8-15,24-31}.  Tile stored [n][k] -> two 16-byte chunks at
// +0 / +32 (+16 for upper lane half).
static __device__ __forceinline__ v16bf ldfragB(const unsigned short* rowbase, int lane) {
    const char* p = (const char*)rowbase + ((lane >> 4) << 4);
    uint8v t;
    t.lo = *(const uint4v*)(p);
    t.hi = *(const uint4v*)(p + 32);
    return __builtin_bit_cast(v16bf, t);
}

// ---------------------------------------------------------------------------
// GEMM1: y1[e][c][h] = relu( sum_m x[e][c][m] * w1[e][h][m] + b1[e][h] ), bf16 out
// grid: (NH/TN, NC/TM, NE), block: 256
// ---------------------------------------------------------------------------
__global__ __launch_bounds__(256)
void moe_gemm1_bf16(const float* __restrict__ x,
                    const float* __restrict__ w1,
                    const float* __restrict__ b1,
                    unsigned short* __restrict__ y1) {
    __shared__ __align__(16) unsigned short lA[TM * LDT];
    __shared__ __align__(16) unsigned short lB[TN * LDT];

    const int e    = blockIdx.z;
    const int c0   = blockIdx.y * TM;
    const int h0   = blockIdx.x * TN;
    const int tid  = threadIdx.x;
    const int lane = tid & 31;
    const int wave = tid >> 5;
    const int wr   = wave >> 1;   // 4 row groups of 32
    const int wc   = wave & 1;    // 2 col groups of 64
    const int lm   = lane & 15;

    const float* gx = x  + (size_t)e * NC * NM + (size_t)c0 * NM;
    const float* gw = w1 + (size_t)e * NH * NM + (size_t)h0 * NM;

    // per-thread staging coordinates (8 float4 per tile per thread)
    int srow[8], skq[8];
#pragma unroll
    for (int r = 0; r < 8; ++r) {
        int idx = tid + r * 256;
        srow[r] = idx >> 4;        // 16 float4 per row
        skq[r]  = idx & 15;
    }

    v8f acc[2][4];
#pragma unroll
    for (int i = 0; i < 2; ++i)
#pragma unroll
        for (int j = 0; j < 4; ++j) acc[i][j] = (v8f)(0.0f);

    float4 pa[8], pb[8];
    // prologue: prefetch tile k0=0
#pragma unroll
    for (int r = 0; r < 8; ++r) {
        pa[r] = *(const float4*)(gx + (size_t)srow[r] * NM + skq[r] * 4);
        pb[r] = *(const float4*)(gw + (size_t)srow[r] * NM + skq[r] * 4);
    }

    for (int k0 = 0; k0 < NM; k0 += TK) {
        // drain staged registers into LDS (fp32 -> bf16)
#pragma unroll
        for (int r = 0; r < 8; ++r) {
            uint2v va = { packbf(pa[r].x, pa[r].y), packbf(pa[r].z, pa[r].w) };
            *(uint2v*)&lA[srow[r] * LDT + skq[r] * 4] = va;
            uint2v vb = { packbf(pb[r].x, pb[r].y), packbf(pb[r].z, pb[r].w) };
            *(uint2v*)&lB[srow[r] * LDT + skq[r] * 4] = vb;
        }
        __syncthreads();

        // prefetch next tile while computing this one
        if (k0 + TK < NM) {
#pragma unroll
            for (int r = 0; r < 8; ++r) {
                pa[r] = *(const float4*)(gx + (size_t)srow[r] * NM + (k0 + TK) + skq[r] * 4);
                pb[r] = *(const float4*)(gw + (size_t)srow[r] * NM + (k0 + TK) + skq[r] * 4);
            }
        }

#pragma unroll
        for (int s = 0; s < TK / KS; ++s) {
            v16bf af[2], bfr[4];
#pragma unroll
            for (int i = 0; i < 2; ++i)
                af[i] = ldfragA(&lA[(wr * 32 + i * 16 + lm) * LDT + s * KS], lane);
#pragma unroll
            for (int j = 0; j < 4; ++j)
                bfr[j] = ldfragB(&lB[(wc * 64 + j * 16 + lm) * LDT + s * KS], lane);
#pragma unroll
            for (int i = 0; i < 2; ++i)
#pragma unroll
                for (int j = 0; j < 4; ++j)
                    acc[i][j] = __builtin_amdgcn_wmma_f32_16x16x32_bf16(
                        false, af[i], false, bfr[j], (short)0, acc[i][j], false, false);
        }
        __syncthreads();   // all waves done reading LDS before next overwrite
    }

    // epilogue: bias + relu + bf16 store. D layout: VGPR jj -> row jj (+8 for
    // upper lane half), lane&15 -> col.
    const int rbase = c0 + wr * 32 + ((lane >> 4) << 3);
#pragma unroll
    for (int j = 0; j < 4; ++j) {
        const int colb = h0 + wc * 64 + j * 16 + lm;
        const float bias = b1[e * NH + colb];
#pragma unroll
        for (int i = 0; i < 2; ++i) {
#pragma unroll
            for (int jj = 0; jj < 8; ++jj) {
                int row = rbase + i * 16 + jj;
                float v = acc[i][j][jj] + bias;
                v = v > 0.0f ? v : 0.0f;
                y1[((size_t)e * NC + row) * NH + colb] = (unsigned short)f2bf1(v);
            }
        }
    }
}

// ---------------------------------------------------------------------------
// GEMM2: out[e][c][o] = sum_h y1[e][c][h] * w2[e][h][o] + b2[e][o], fp32 out
// grid: (NO/TN, NC/TM, NE), block: 256
// ---------------------------------------------------------------------------
__global__ __launch_bounds__(256)
void moe_gemm2_bf16(const unsigned short* __restrict__ y1,
                    const float* __restrict__ w2,
                    const float* __restrict__ b2,
                    float* __restrict__ out) {
    __shared__ __align__(16) unsigned short lA[TM * LDT];
    __shared__ __align__(16) unsigned short lB[TN * LDT];

    const int e    = blockIdx.z;
    const int c0   = blockIdx.y * TM;
    const int o0   = blockIdx.x * TN;
    const int tid  = threadIdx.x;
    const int lane = tid & 31;
    const int wave = tid >> 5;
    const int wr   = wave >> 1;
    const int wc   = wave & 1;
    const int lm   = lane & 15;

    const unsigned short* gy = y1 + (size_t)e * NC * NH + (size_t)c0 * NH;
    const float*          gw = w2 + (size_t)e * NH * NO + o0;

    // A staging: 4 x 16B chunks/thread; B staging: 16 packed dwords/thread
    int arow[4], aq[4], bo[16], bhp[16];
#pragma unroll
    for (int r = 0; r < 4; ++r) {
        int idx = tid + r * 256;
        arow[r] = idx >> 3;        // 8 chunks per row
        aq[r]   = idx & 7;
    }
#pragma unroll
    for (int r = 0; r < 16; ++r) {
        int idx = tid + r * 256;   // 4096 dwords total
        bo[r]  = idx & 127;
        bhp[r] = idx >> 7;         // 0..31
    }

    v8f acc[2][4];
#pragma unroll
    for (int i = 0; i < 2; ++i)
#pragma unroll
        for (int j = 0; j < 4; ++j) acc[i][j] = (v8f)(0.0f);

    uint4v qa[4];
    float  pb0[16], pb1[16];
    // prologue: prefetch tile k0=0
#pragma unroll
    for (int r = 0; r < 4; ++r)
        qa[r] = *(const uint4v*)(gy + (size_t)arow[r] * NH + aq[r] * 8);
#pragma unroll
    for (int r = 0; r < 16; ++r) {
        const float* p = gw + (size_t)(2 * bhp[r]) * NO + bo[r];
        pb0[r] = p[0];
        pb1[r] = p[NO];
    }

    for (int k0 = 0; k0 < NH; k0 += TK) {
        // drain staged registers into LDS (A: raw bf16 copy; B: transpose+cvt)
#pragma unroll
        for (int r = 0; r < 4; ++r)
            *(uint4v*)&lA[arow[r] * LDT + aq[r] * 8] = qa[r];
#pragma unroll
        for (int r = 0; r < 16; ++r)
            *(unsigned*)&lB[bo[r] * LDT + 2 * bhp[r]] = packbf(pb0[r], pb1[r]);
        __syncthreads();

        // prefetch next tile while computing this one
        if (k0 + TK < NH) {
#pragma unroll
            for (int r = 0; r < 4; ++r)
                qa[r] = *(const uint4v*)(gy + (size_t)arow[r] * NH + (k0 + TK) + aq[r] * 8);
#pragma unroll
            for (int r = 0; r < 16; ++r) {
                const float* p = gw + (size_t)(k0 + TK + 2 * bhp[r]) * NO + bo[r];
                pb0[r] = p[0];
                pb1[r] = p[NO];
            }
        }

#pragma unroll
        for (int s = 0; s < TK / KS; ++s) {
            v16bf af[2], bfr[4];
#pragma unroll
            for (int i = 0; i < 2; ++i)
                af[i] = ldfragA(&lA[(wr * 32 + i * 16 + lm) * LDT + s * KS], lane);
#pragma unroll
            for (int j = 0; j < 4; ++j)
                bfr[j] = ldfragB(&lB[(wc * 64 + j * 16 + lm) * LDT + s * KS], lane);
#pragma unroll
            for (int i = 0; i < 2; ++i)
#pragma unroll
                for (int j = 0; j < 4; ++j)
                    acc[i][j] = __builtin_amdgcn_wmma_f32_16x16x32_bf16(
                        false, af[i], false, bfr[j], (short)0, acc[i][j], false, false);
        }
        __syncthreads();
    }

    const int rbase = c0 + wr * 32 + ((lane >> 4) << 3);
#pragma unroll
    for (int j = 0; j < 4; ++j) {
        const int colb = o0 + wc * 64 + j * 16 + lm;
        const float bias = b2[e * NO + colb];
#pragma unroll
        for (int i = 0; i < 2; ++i) {
#pragma unroll
            for (int jj = 0; jj < 8; ++jj) {
                int row = rbase + i * 16 + jj;
                out[((size_t)e * NC + row) * NO + colb] = acc[i][j][jj] + bias;
            }
        }
    }
}

// ---------------------------------------------------------------------------
extern "C" void kernel_launch(void* const* d_in, const int* in_sizes, int n_in,
                              void* d_out, int out_size, void* d_ws, size_t ws_size,
                              hipStream_t stream) {
    const float* x     = (const float*)d_in[0];   // (E, C, M)
    const float* fc1_w = (const float*)d_in[1];   // (E, H, M)
    const float* fc1_b = (const float*)d_in[2];   // (E, H)
    const float* fc2_w = (const float*)d_in[3];   // (E, H, O)
    const float* fc2_b = (const float*)d_in[4];   // (E, O)
    float* out = (float*)d_out;                   // (E, C, O)
    unsigned short* y1 = (unsigned short*)d_ws;   // (E, C, H) bf16, 128 MB

    dim3 g1(NH / TN, NC / TM, NE);
    dim3 g2(NO / TN, NC / TM, NE);
    moe_gemm1_bf16<<<g1, 256, 0, stream>>>(x, fc1_w, fc1_b, y1);
    moe_gemm2_bf16<<<g2, 256, 0, stream>>>(y1, fc2_w, fc2_b, out);
}
